// MSDeformAttn_23261542875599
// MI455X (gfx1250) — matrix operand order repaired and checked by reading, not compiled, not run on hardware
//
#include <hip/hip_runtime.h>
#include <hip/hip_bf16.h>

typedef __attribute__((ext_vector_type(16))) _Float16     v16h;
typedef __attribute__((ext_vector_type(8)))  float        v8f;
typedef __attribute__((ext_vector_type(2)))  float        v2f;
typedef __attribute__((ext_vector_type(4)))  unsigned int v4u;
typedef __attribute__((ext_vector_type(8)))  int          v8i;
typedef __attribute__((ext_vector_type(4)))  int          v4i;

#define D_MODEL  256
#define N_HEADS  26
#define N_POINTS 4
#define ACTIVE   8
#define K_NN     3
#define D_HEAD   32
#define NQ       2048
#define NSRC     2048

#if defined(__has_builtin)
#if __has_builtin(__builtin_amdgcn_tensor_load_to_lds)
#define USE_TDM 1
#endif
#endif
#ifndef USE_TDM
#define USE_TDM 0
#endif

#if USE_TDM
// ---------------------------------------------------------------------------
// TDM 2D tile load: (tile_dim1=32 rows) x (tile_dim0=16 cols) of f32 from a
// row-major tensor with row stride `stride_elems`, into LDS at `lds_off`.
// OOB columns/rows (beyond cols_valid/rows_valid) read as zero per ISA 8.2.
// Descriptor packing per CDNA5 ISA 8.3/8.4 (D# group 0 + group 1).
// 6-arg builtin form (clang-23 / therock-10.0 headers): groups 2,3 unused
// for 2D tensors (VADDR2/3 NULL semantics), trailing int32x8 zeroed.
// ---------------------------------------------------------------------------
__device__ __forceinline__ void tdm_load_tile32x16(
    unsigned lds_off, const float* gaddr,
    unsigned cols_valid, unsigned rows_valid, unsigned stride_elems)
{
  const unsigned long long ga = (unsigned long long)(uintptr_t)gaddr;
  v4u g0;
  g0[0] = 1u;                                      // count=1 (valid user D#)
  g0[1] = lds_off;                                 // lds_addr [63:32]
  g0[2] = (unsigned)(ga & 0xffffffffu);            // global_addr [95:64]
  g0[3] = (unsigned)((ga >> 32) & 0x01ffffffu)     // global_addr [120:96]
        | (2u << 30);                              // type=2 ("image") [127:126]
  v8i g1;
  g1[0] = (int)(2u << 16);                         // data_size=4B; wg_mask=0
  g1[1] = (int)((cols_valid & 0xffffu) << 16);     // tensor_dim0[15:0] @ [63:48]
  g1[2] = (int)((cols_valid >> 16) |               // tensor_dim0[31:16]
                ((rows_valid & 0xffffu) << 16));   // tensor_dim1[15:0]
  g1[3] = (int)((rows_valid >> 16) |               // tensor_dim1[31:16]
                (16u << 16));                      // tile_dim0 = 16
  g1[4] = 32;                                      // tile_dim1 = 32, tile_dim2 = 0
  g1[5] = (int)stride_elems;                       // tensor_dim0_stride[31:0]
  g1[6] = 0;                                       // stride[47:32]=0, dim1_stride lo=0
  g1[7] = 0;
  const v4i z4 = {0, 0, 0, 0};
  const v8i z8 = {0, 0, 0, 0, 0, 0, 0, 0};
  __builtin_amdgcn_tensor_load_to_lds(g0, g1, z4, z4, z8, 0);
}
#endif

// ---------------------------------------------------------------------------
// Generic GEMM: C[M,N] = A[M,K] @ W[K,N] + bias[N].  f32 in/out, f16 WMMA core.
// One wave32 computes one 16x16 tile of C via v_wmma_f32_16x16x32_f16.
// W k-slabs (32x16) staged into per-wave LDS double buffers by the TDM,
// overlapped with WMMA on the previous slab (s_wait_tensorcnt pipelining).
// Fragment layouts per CDNA5 ISA 7.12.2 (wave32).
// ---------------------------------------------------------------------------
__global__ __launch_bounds__(256) void gemm_wmma_f16(
    const float* __restrict__ A, const float* __restrict__ W,
    const float* __restrict__ bias, float* __restrict__ C,
    int M, int N, int K, int ntn)
{
#if USE_TDM
  __shared__ float wtile[8][2][32 * 16];           // 32 KB: 8 waves x 2 bufs
#endif
  const int lane = threadIdx.x & 31;
  const int wv   = threadIdx.x >> 5;
  const int gw   = blockIdx.x * (blockDim.x >> 5) + wv;
  const int tmTot = M >> 4;
  if (gw >= tmTot * ntn) return;
  const int tm = gw / ntn;
  const int tn = gw % ntn;
  const int r  = lane & 15;          // A-row within tile / C-col within tile
  const int h  = lane >> 4;          // half selector
  const int col = tn * 16 + r;       // global output column
  const bool cin = (col < N);

  v8f acc = {};
  const float* arow = A + (size_t)(tm * 16 + r) * K;

#if USE_TDM
  const unsigned lds0 = (unsigned)(uintptr_t)&wtile[wv][0][0];
  const unsigned lds1 = (unsigned)(uintptr_t)&wtile[wv][1][0];
  const unsigned remCols = (unsigned)(N - tn * 16);
  tdm_load_tile32x16(lds0, W + tn * 16, remCols, (unsigned)K, (unsigned)N);
#else
  const int colc = col < N ? col : (N - 1);        // branchless clamp
#endif

  for (int k0 = 0; k0 < K; k0 += 32) {
    if (k0 + 32 < K)
      __builtin_prefetch(arow + k0 + 32, 0, 1);    // global_prefetch_b8

#if USE_TDM
    const int cur = (k0 >> 5) & 1;
    if (k0 + 32 < K) {
      tdm_load_tile32x16(cur ? lds0 : lds1, W + (size_t)(k0 + 32) * N + tn * 16,
                         remCols, (unsigned)(K - (k0 + 32)), (unsigned)N);
      __builtin_amdgcn_s_wait_tensorcnt(1);        // current slab resident
    } else {
      __builtin_amdgcn_s_wait_tensorcnt(0);
    }
    const float* wl = &wtile[wv][cur][0];
#endif

    v16h a, b;
#pragma unroll
    for (int i = 0; i < 8; ++i) {
      const int ka = k0 + h * 8 + i;               // A: interleaved K halves
      a[i]     = (_Float16)arow[ka];
      a[i + 8] = (_Float16)arow[ka + 16];
    }
#pragma unroll
    for (int i = 0; i < 16; ++i) {                 // B: interleaved K halves
#if USE_TDM
      b[i] = (_Float16)wl[(h * 8 + (i & 7) + ((i >> 3) << 4)) * 16 + r];  // ds_load, OOB already 0
#else
      const float wvv = W[(size_t)(k0 + h * 8 + (i & 7) + ((i >> 3) << 4)) * N + colc];
      b[i] = (_Float16)(cin ? wvv : 0.f);
#endif
    }
    acc = __builtin_amdgcn_wmma_f32_16x16x32_f16(
        /*neg_a=*/false, a, /*neg_b=*/false, b,
        /*c_mod=*/(short)0, acc, /*reuse_a=*/false, /*reuse_b=*/false);
  }

  if (cin) {
    const float bv = bias ? bias[col] : 0.f;
#pragma unroll
    for (int j = 0; j < 8; ++j) {
      const int row = tm * 16 + j + 8 * h;
      C[(size_t)row * N + col] = acc[j] + bv;
    }
  }
}

// ---------------------------------------------------------------------------
// Normalize source coords and precompute |m|^2 -> float4 {x,y,z,|m|^2}.
// ---------------------------------------------------------------------------
__global__ void prep_coords(const float* __restrict__ coords,
                            const float* __restrict__ sr,
                            float4* __restrict__ mc)
{
  const int m = blockIdx.x * blockDim.x + threadIdx.x;
  if (m >= NSRC) return;
  const float x = (coords[m * 3 + 0] - sr[0]) / (sr[3] - sr[0] + 1e-7f);
  const float y = (coords[m * 3 + 1] - sr[1]) / (sr[4] - sr[1] + 1e-7f);
  const float z = (coords[m * 3 + 2] - sr[2]) / (sr[5] - sr[2] + 1e-7f);
  mc[m] = make_float4(x, y, z, x * x + y * y + z * z);
}

// ---------------------------------------------------------------------------
// Per-query: head scores, top-8 heads, softmax over points, normalized
// sampling locations.  One thread per query.
// ---------------------------------------------------------------------------
__global__ void topk_select(const float* __restrict__ aw,
                            const float* __restrict__ so,
                            const float* __restrict__ ref,
                            const float* __restrict__ sr,
                            int*   __restrict__ topk,
                            float* __restrict__ awsel,
                            float* __restrict__ locs)
{
  const int q = blockIdx.x * blockDim.x + threadIdx.x;
  if (q >= NQ) return;
  const float* awq = aw + (size_t)q * (N_HEADS * N_POINTS);
  const float* soq = so + (size_t)q * (N_HEADS * N_POINTS * 3);
  const float* rq  = ref + (size_t)q * 3;

  float s[N_HEADS];
#pragma unroll 1
  for (int hh = 0; hh < N_HEADS; ++hh)
    s[hh] = awq[hh * 4 + 0] + awq[hh * 4 + 1] + awq[hh * 4 + 2] + awq[hh * 4 + 3];

  float mn[3], dn[3];
#pragma unroll
  for (int c = 0; c < 3; ++c) { mn[c] = sr[c]; dn[c] = sr[3 + c] - sr[c] + 1e-7f; }

  for (int e = 0; e < ACTIVE; ++e) {
    int best = 0; float bv = -3.4e38f;
#pragma unroll 1
    for (int hh = 0; hh < N_HEADS; ++hh)
      if (s[hh] > bv) { bv = s[hh]; best = hh; }
    s[best] = -3.4e38f;
    topk[q * ACTIVE + e] = best;

    float v[4], mx = -3.4e38f;
#pragma unroll
    for (int p = 0; p < 4; ++p) { v[p] = awq[best * 4 + p]; mx = fmaxf(mx, v[p]); }
    float sum = 0.f;
#pragma unroll
    for (int p = 0; p < 4; ++p) { v[p] = __expf(v[p] - mx); sum += v[p]; }
#pragma unroll
    for (int p = 0; p < 4; ++p)
      awsel[((size_t)q * ACTIVE + e) * 4 + p] = v[p] / sum;

#pragma unroll
    for (int p = 0; p < 4; ++p) {
      const float* sp = soq + (best * 4 + p) * 3;
      float* lp = locs + (((size_t)q * ACTIVE + e) * 4 + p) * 3;
#pragma unroll
      for (int c = 0; c < 3; ++c)
        lp[c] = (rq[c] + sp[c] - mn[c]) / dn[c];
    }
  }
}

// ---------------------------------------------------------------------------
// Gather active-head value rows: lv[e][n][d] = value[n][topk[n][e]*32 + d]
// ---------------------------------------------------------------------------
__global__ void gather_value(const float* __restrict__ value,
                             const int* __restrict__ topk,
                             float* __restrict__ lv)
{
  const int t = blockIdx.x * blockDim.x + threadIdx.x;
  if (t >= ACTIVE * NQ * D_HEAD) return;
  const int d = t & 31;
  const int n = (t >> 5) & (NQ - 1);
  const int e = t >> 16;
  const int hsel = topk[n * ACTIVE + e];
  lv[t] = value[(size_t)n * (N_HEADS * D_HEAD) + hsel * D_HEAD + d];
}

// ---------------------------------------------------------------------------
// kNN core.  One wave32 per (e, 16 query-points).  Distance tiles computed
// with V_WMMA_F32_16X16X4_F32:
//   A row m   = [mx, my, mz, |m|^2]           (f32 A 16x4 layout, ISA 7.12.2)
//   B col qp  = [-2qx, -2qy, -2qz, 1]
//   C seeded with |q|^2  ->  D[m,qp] = ||m - q||^2   exactly.
// Source coords staged once per block into LDS (32 KB of the 320 KB WGP LDS),
// so the 128-iteration stream runs out of ds_load_b128, not vmem.
// ---------------------------------------------------------------------------
__global__ __launch_bounds__(256) void knn_core(
    const float4* __restrict__ mc, const float* __restrict__ locs,
    const float* __restrict__ lv, const float* __restrict__ awsel,
    float* __restrict__ outc)
{
  __shared__ float4 smc[NSRC];                      // 32 KB
  for (int t = threadIdx.x; t < NSRC; t += blockDim.x) smc[t] = mc[t];
  __syncthreads();

  const int lane = threadIdx.x & 31;
  const int gw   = blockIdx.x * (blockDim.x >> 5) + (threadIdx.x >> 5);
  const int e    = gw >> 9;          // 512 tiles per e
  const int tile = gw & 511;
  const int cidx = lane & 15;
  const int h    = lane >> 4;

  const int j = tile * 16 + cidx;    // flattened (q,p)
  const int q = j >> 2;
  const int p = j & 3;

  const float* L = locs + (((size_t)q * ACTIVE + e) * 4 + p) * 3;
  const float qx = L[0], qy = L[1], qz = L[2];
  const float qn2 = qx * qx + qy * qy + qz * qz;
  const float bq[4] = { -2.f * qx, -2.f * qy, -2.f * qz, 1.f };
  v2f bfrag; bfrag[0] = bq[2 * h]; bfrag[1] = bq[2 * h + 1];

  float d0 = 3.4e38f, d1 = 3.4e38f, d2v = 3.4e38f;
  int   i0 = 0, i1 = 0, i2 = 0;

  for (int mt = 0; mt < NSRC / 16; ++mt) {
    const float4 m4 = smc[mt * 16 + cidx];
    const float av[4] = { m4.x, m4.y, m4.z, m4.w };
    v2f afrag; afrag[0] = av[2 * h]; afrag[1] = av[2 * h + 1];

    v8f c;
#pragma unroll
    for (int r = 0; r < 8; ++r) c[r] = qn2;
    v8f dmat = __builtin_amdgcn_wmma_f32_16x16x4_f32(
        false, afrag, false, bfrag, (short)0, c, false, false);

#pragma unroll
    for (int r = 0; r < 8; ++r) {
      const float val = dmat[r];
      const int midx = mt * 16 + r + 8 * h;
      if (val < d0)      { d2v = d1; i2 = i1; d1 = d0; i1 = i0; d0 = val; i0 = midx; }
      else if (val < d1) { d2v = d1; i2 = i1; d1 = val; i1 = midx; }
      else if (val < d2v){ d2v = val; i2 = midx; }
    }
  }

  // merge with the partner lane holding the other 8 rows of the same column
  {
    float od[3]; int oi[3];
    od[0] = __shfl_xor(d0, 16, 32);  oi[0] = __shfl_xor(i0, 16, 32);
    od[1] = __shfl_xor(d1, 16, 32);  oi[1] = __shfl_xor(i1, 16, 32);
    od[2] = __shfl_xor(d2v, 16, 32); oi[2] = __shfl_xor(i2, 16, 32);
#pragma unroll
    for (int t = 0; t < 3; ++t) {
      const float val = od[t]; const int midx = oi[t];
      if (val < d0)      { d2v = d1; i2 = i1; d1 = d0; i1 = i0; d0 = val; i0 = midx; }
      else if (val < d1) { d2v = d1; i2 = i1; d1 = val; i1 = midx; }
      else if (val < d2v){ d2v = val; i2 = midx; }
    }
  }

  // inverse-distance weights (reference: 1/(dist+1e-7), normalized)
  const float t0 = sqrtf(fmaxf(d0, 0.f));
  const float t1 = sqrtf(fmaxf(d1, 0.f));
  const float t2 = sqrtf(fmaxf(d2v, 0.f));
  float w0 = 1.f / (t0 + 1e-7f), w1 = 1.f / (t1 + 1e-7f), w2 = 1.f / (t2 + 1e-7f);
  const float ws = w0 + w1 + w2;
  w0 /= ws; w1 /= ws; w2 /= ws;

  const float awv = awsel[((size_t)q * ACTIVE + e) * 4 + p];
  const float* base = lv + (size_t)e * NQ * D_HEAD;
  const float* f0 = base + (size_t)i0 * D_HEAD;
  const float* f1 = base + (size_t)i1 * D_HEAD;
  const float* f2 = base + (size_t)i2 * D_HEAD;

#pragma unroll 4
  for (int d = 0; d < D_HEAD; ++d) {
    float v = awv * (w0 * f0[d] + w1 * f1[d] + w2 * f2[d]);
    v += __shfl_xor(v, 1, 32);       // sum the 4 point-lanes of this query
    v += __shfl_xor(v, 2, 32);
    if (h == 0 && p == 0)
      outc[(size_t)q * D_MODEL + e * D_HEAD + d] = v;
  }
}

// ---------------------------------------------------------------------------
// Launch
// ---------------------------------------------------------------------------
extern "C" void kernel_launch(void* const* d_in, const int* in_sizes, int n_in,
                              void* d_out, int out_size, void* d_ws, size_t ws_size,
                              hipStream_t stream)
{
  const float* query   = (const float*)d_in[0];
  const float* coords  = (const float*)d_in[1];
  const float* sr      = (const float*)d_in[2];
  const float* ref     = (const float*)d_in[3];
  const float* xflat   = (const float*)d_in[4];
  const float* W_so    = (const float*)d_in[5];
  const float* b_so    = (const float*)d_in[6];
  const float* W_aw    = (const float*)d_in[7];
  const float* b_aw    = (const float*)d_in[8];
  const float* W_v     = (const float*)d_in[9];
  const float* b_v     = (const float*)d_in[10];
  const float* W_out   = (const float*)d_in[11];
  const float* b_out   = (const float*)d_in[12];
  float* out = (float*)d_out;

  char* ws = (char*)d_ws;
  float*  aw    = (float*)(ws + 0);                       // 2048 x 104
  float*  so    = (float*)(ws + 851968);                  // 2048 x 312
  float*  value = (float*)(ws + 3407872);                 // 2048 x 832
  float*  lv    = (float*)(ws + 10223616);                // 8 x 2048 x 32
  int*    topk  = (int*)  (ws + 12320768);                // 2048 x 8
  float*  awsel = (float*)(ws + 12386304);                // 2048 x 8 x 4
  float*  locs  = (float*)(ws + 12648448);                // 2048 x 8 x 4 x 3
  float4* mcrd  = (float4*)(ws + 13434880);               // 2048 x 4
  float*  core  = (float*)(ws + 13467648);                // 2048 x 256

  // Projections (WMMA f16 GEMMs, TDM-staged weight tiles)
  {
    int ntn = (N_HEADS * N_POINTS + 15) / 16;             // 7
    int waves = (NQ / 16) * ntn;
    gemm_wmma_f16<<<(waves + 7) / 8, 256, 0, stream>>>(
        query, W_aw, b_aw, aw, NQ, N_HEADS * N_POINTS, D_MODEL, ntn);
  }
  {
    int ntn = (N_HEADS * N_POINTS * 3 + 15) / 16;         // 20
    int waves = (NQ / 16) * ntn;
    gemm_wmma_f16<<<(waves + 7) / 8, 256, 0, stream>>>(
        query, W_so, b_so, so, NQ, N_HEADS * N_POINTS * 3, D_MODEL, ntn);
  }
  {
    int ntn = (N_HEADS * D_HEAD + 15) / 16;               // 52
    int waves = (NQ / 16) * ntn;
    gemm_wmma_f16<<<(waves + 7) / 8, 256, 0, stream>>>(
        xflat, W_v, b_v, value, NQ, N_HEADS * D_HEAD, D_MODEL, ntn);
  }

  prep_coords<<<NSRC / 256, 256, 0, stream>>>(coords, sr, mcrd);
  topk_select<<<NQ / 256, 256, 0, stream>>>(aw, so, ref, sr, topk, awsel, locs);
  gather_value<<<(ACTIVE * NQ * D_HEAD) / 256, 256, 0, stream>>>(value, topk, lv);

  // kNN + interpolate + point-weighted sum (WMMA f32 distance tiles)
  knn_core<<<(ACTIVE * ((NQ * N_POINTS) / 16)) / 8, 256, 0, stream>>>(
      mcrd, locs, lv, awsel, core);

  // Output projection
  {
    int ntn = D_MODEL / 16;                               // 16
    int waves = (NQ / 16) * ntn;
    gemm_wmma_f16<<<(waves + 7) / 8, 256, 0, stream>>>(
        core, W_out, b_out, out, NQ, D_MODEL, D_MODEL, ntn);
  }
}
